// LocalExperts_23278722744539
// MI455X (gfx1250) — matrix operand order repaired and checked by reading, not compiled, not run on hardware
//
#include <hip/hip_runtime.h>
#include <stdint.h>

// ---------------- problem constants ----------------
#define E_  8
#define D_  1024
#define F_  4096
#define T_  16384
#define C_  (T_ / E_)     // 2048 tokens per expert
#define TMS 64            // output rows per workgroup (swizzled fast path)
#define TMF 32            // output rows per workgroup (fallback path)
#define FK  64            // F-chunk per fused step

// Swizzled fragment tile: 16x32 (A) or 32x16 (B) elements -> 512 dwords.
//   dwords [lane*8 .. lane*8+7]        : hi-plane (8 dwords = 16 bf16, K-pairs)
//   dwords [256+lane*8 .. 256+lane*8+7]: lo-plane
#define TILE_DW 512

#define XS_DW   ((size_t)(T_ / 16) * (D_ / 32) * TILE_DW)        // x  tiles
#define W1S_DW  ((size_t)E_ * (D_ / 32) * (F_ / 16) * TILE_DW)   // w1 tiles
#define W2S_DW  ((size_t)E_ * (F_ / 32) * (D_ / 16) * TILE_DW)   // w2 tiles
#define WS_NEED_BYTES ((XS_DW + W1S_DW + W2S_DW) * 4)

typedef __attribute__((ext_vector_type(16))) __bf16 v16bf;
typedef __attribute__((ext_vector_type(8)))  float  v8f;

union V16 { v16bf v; uint32_t u[8]; };

// round-to-nearest-even fp32 -> bf16 (low 16 bits)
__device__ __forceinline__ uint32_t bf16_rne(float x) {
    uint32_t u = __float_as_uint(x);
    return (u + 0x7fffu + ((u >> 16) & 1u)) >> 16;
}

// split fp32 into (hi,lo) bf16 packed in one dword: [31:16]=lo [15:0]=hi
__device__ __forceinline__ uint32_t pack_split(float x) {
    uint32_t h = bf16_rne(x);
    float hf = __uint_as_float(h << 16);
    uint32_t l = bf16_rne(x - hf);
    return (l << 16) | h;
}

__device__ __forceinline__ void split2(float f0, float f1, uint32_t& hi, uint32_t& lo) {
    uint32_t h0 = bf16_rne(f0), h1 = bf16_rne(f1);
    float r0 = f0 - __uint_as_float(h0 << 16);
    float r1 = f1 - __uint_as_float(h1 << 16);
    hi = (h1 << 16) | h0;
    lo = (bf16_rne(r1) << 16) | bf16_rne(r0);
}

__device__ __forceinline__ float gelu_tanh(float x) {
    const float c0 = 0.7978845608028654f;   // sqrt(2/pi)
    const float c1 = 0.044715f;
    float x3 = x * x * x;
    return 0.5f * x * (1.0f + tanhf(c0 * (x + c1 * x3)));
}

__device__ __forceinline__ v8f wmma_bf16(const V16& a, const V16& b, v8f c) {
    return __builtin_amdgcn_wmma_f32_16x16x32_bf16(false, a.v, false, b.v,
                                                   (short)0, c, false, false);
}

// ---- load a pre-swizzled fragment: pure b128 loads, no conversion VALU ----
__device__ __forceinline__ void load_frag(const uint32_t* __restrict__ tile, int lane,
                                          V16& h, V16& l) {
    const uint4* ph = (const uint4*)(tile + lane * 8);
    const uint4* pl = (const uint4*)(tile + 256 + lane * 8);
    uint4 a = ph[0], b = ph[1];
    uint4 c = pl[0], d = pl[1];
    h.u[0] = a.x; h.u[1] = a.y; h.u[2] = a.z; h.u[3] = a.w;
    h.u[4] = b.x; h.u[5] = b.y; h.u[6] = b.z; h.u[7] = b.w;
    l.u[0] = c.x; l.u[1] = c.y; l.u[2] = c.z; l.u[3] = c.w;
    l.u[4] = d.x; l.u[5] = d.y; l.u[6] = d.z; l.u[7] = d.w;
}

// ---- A fragment from LDS of packed split-bf16 dwords ----
// A layout: lanes 0-15: M=lane, K={0..7,16..23}; lanes 16-31: M=lane-16, K={8..15,24..31}
__device__ __forceinline__ void load_a_lds(const uint32_t* __restrict__ src, int ld, int k0,
                                           int lane, V16& ah, V16& al) {
    const int m  = lane & 15;
    const int kh = (lane >> 4) << 3;
    const uint32_t* row = src + m * ld + k0;
#pragma unroll
    for (int j = 0; j < 8; ++j) {
        int koff = ((j & 4) << 2) + kh + ((j & 3) << 1);
        uint32_t u0 = row[koff];
        uint32_t u1 = row[koff + 1];
        ah.u[j] = ((u1 & 0xffffu) << 16) | (u0 & 0xffffu);
        al.u[j] = (u1 & 0xffff0000u) | (u0 >> 16);
    }
}

// ================= prep kernels: split + fragment-swizzle =================

// x [T x D] row-major -> A-fragment tiles (rowtile16, ktile32)
__global__ void swizzle_x_kernel(const float* __restrict__ x, uint32_t* __restrict__ xs) {
    size_t idx = (size_t)blockIdx.x * blockDim.x + threadIdx.x;
    const size_t ntiles = (size_t)(T_ / 16) * (D_ / 32);
    if (idx >= ntiles * 256) return;
    size_t tile = idx >> 8;
    int lj = (int)(idx & 255);
    int lane = lj >> 3, j = lj & 7;
    int kt = (int)(tile % (D_ / 32));
    int rt = (int)(tile / (D_ / 32));
    int m  = lane & 15;
    int kh = (lane >> 4) << 3;
    int koff = ((j & 4) << 2) + kh + ((j & 3) << 1);
    const float* src = x + (size_t)(rt * 16 + m) * D_ + kt * 32 + koff;
    uint32_t hi, lo;
    split2(src[0], src[1], hi, lo);
    uint32_t* out = xs + tile * TILE_DW + lane * 8 + j;
    out[0]   = hi;
    out[256] = lo;
}

// w [E x K x N] row-major -> B-fragment tiles (e, ktile32, ntile16)
// B layout: lanes 0-15: N=lane, K={0..7,16..23}; lanes 16-31: N=lane-16, K={8..15,24..31} (K pairs per dword)
__global__ void swizzle_w_kernel(const float* __restrict__ w, uint32_t* __restrict__ out_ws,
                                 int K, int N) {
    size_t idx = (size_t)blockIdx.x * blockDim.x + threadIdx.x;
    const size_t tiles_per_e = (size_t)(K / 32) * (N / 16);
    if (idx >= (size_t)E_ * tiles_per_e * 256) return;
    size_t t = idx >> 8;
    int lj = (int)(idx & 255);
    int lane = lj >> 3, j = lj & 7;
    int nt = (int)(t % (N / 16));
    size_t t2 = t / (N / 16);
    int kt = (int)(t2 % (K / 32));
    int e  = (int)(t2 / (K / 32));
    int n  = nt * 16 + (lane & 15);
    int k  = kt * 32 + ((j & 4) << 2) + ((lane >> 4) << 3) + ((j & 3) << 1);
    const float* src = w + ((size_t)e * K + k) * N + n;
    uint32_t hi, lo;
    split2(src[0], src[N], hi, lo);
    uint32_t* out = out_ws + t * TILE_DW + lane * 8 + j;
    out[0]   = hi;
    out[256] = lo;
}

// ================= fused FFN on pre-swizzled operands (TM=64) =================
__global__ __launch_bounds__(256, 1)
void moe_ffn_fused_sw(const uint32_t* __restrict__ xs,
                      const uint32_t* __restrict__ w1s,
                      const float*    __restrict__ b1,
                      const uint32_t* __restrict__ w2s,
                      const float*    __restrict__ b2,
                      float*          __restrict__ y)
{
    __shared__ uint32_t lds_h[TMS * FK];    // split-bf16 packed GELU activations (16 KB)

    const int wg      = blockIdx.x;
    const int e       = wg >> 5;            // 32 row tiles per expert
    const int rt      = wg & 31;
    const int rowbase = e * C_ + rt * TMS;
    const int row16   = rowbase >> 4;       // global 16-row tile index

    const int tid  = threadIdx.x;
    const int lane = tid & 31;
    const int w    = tid >> 5;

    const float* b1e = b1 + (size_t)e * F_;
    const float* b2e = b2 + (size_t)e * D_;

    const int hr0   = (w >> 2) << 1;        // GEMM1 h-tile rows hr0, hr0+1 (0..3)
    const int hc    = w & 3;                // GEMM1 h-tile col (0..3)
    const int nbase = w * 128;              // GEMM2 column ownership

    v8f acc[4][8];
#pragma unroll
    for (int r = 0; r < 4; ++r)
#pragma unroll
        for (int t = 0; t < 8; ++t)
            acc[r][t] = (v8f){0.f, 0.f, 0.f, 0.f, 0.f, 0.f, 0.f, 0.f};

    // base tile pointers
    const uint32_t* xa0 = xs + (size_t)(row16 + hr0)     * (D_ / 32) * TILE_DW;
    const uint32_t* xa1 = xs + (size_t)(row16 + hr0 + 1) * (D_ / 32) * TILE_DW;
    const uint32_t* w1e_base = w1s + (size_t)e * (D_ / 32) * (F_ / 16) * TILE_DW;
    const uint32_t* w2e_base = w2s + (size_t)e * (F_ / 32) * (D_ / 16) * TILE_DW;

#pragma unroll 1
    for (int fc = 0; fc < F_ / FK; ++fc) {
        const int f0 = fc * FK;

        if (fc + 1 < F_ / FK)
            __builtin_prefetch(w2e_base + (size_t)((fc + 1) * 2) * (D_ / 16) * TILE_DW
                               + (nbase >> 4) * TILE_DW + lane * 8, 0, 1);

        // ---- GEMM1: h(64x64) = x_tile @ w1[:, f0:f0+64]
        //      each wave: 2 row tiles x 1 col tile, one shared B fragment, 6 chains ----
        v8f p0 = (v8f){0.f, 0.f, 0.f, 0.f, 0.f, 0.f, 0.f, 0.f};
        v8f p1 = p0, p2 = p0, p3 = p0, p4 = p0, p5 = p0;
#pragma unroll 4
        for (int ks = 0; ks < D_ / 32; ++ks) {
            V16 a0h, a0l, a1h, a1l, bh, bl;
            load_frag(xa0 + (size_t)ks * TILE_DW, lane, a0h, a0l);
            load_frag(xa1 + (size_t)ks * TILE_DW, lane, a1h, a1l);
            load_frag(w1e_base + ((size_t)ks * (F_ / 16) + (f0 >> 4) + hc) * TILE_DW,
                      lane, bh, bl);
            p0 = wmma_bf16(a0h, bh, p0);
            p3 = wmma_bf16(a1h, bh, p3);
            p1 = wmma_bf16(a0h, bl, p1);
            p4 = wmma_bf16(a1h, bl, p4);
            p2 = wmma_bf16(a0l, bh, p2);
            p5 = wmma_bf16(a1l, bh, p5);
        }
        v8f hacc0 = p0 + p1 + p2;
        v8f hacc1 = p3 + p4 + p5;

        // bias + GELU + split-pack into LDS (both row tiles)
        {
            const float bv  = b1e[f0 + hc * 16 + (lane & 15)];
            const int   col = hc * 16 + (lane & 15);
            const int   m0  = (hr0)     * 16 + ((lane >> 4) << 3);
            const int   m1  = (hr0 + 1) * 16 + ((lane >> 4) << 3);
#pragma unroll
            for (int v = 0; v < 8; ++v) {
                lds_h[(m0 + v) * FK + col] = pack_split(gelu_tanh(hacc0[v] + bv));
                lds_h[(m1 + v) * FK + col] = pack_split(gelu_tanh(hacc1[v] + bv));
            }
        }
        __syncthreads();

        // ---- GEMM2: y(64x1024) += h(64x64) @ w2[f0:f0+64, :]
        //      each B fragment feeds 4 row tiles (12 WMMAs) ----
#pragma unroll
        for (int kk = 0; kk < 2; ++kk) {
            V16 Ah[4], Al[4];
#pragma unroll
            for (int r = 0; r < 4; ++r)
                load_a_lds(lds_h + r * 16 * FK, FK, kk * 32, lane, Ah[r], Al[r]);
            const uint32_t* w2k = w2e_base + (size_t)(fc * 2 + kk) * (D_ / 16) * TILE_DW;
#pragma unroll
            for (int tc = 0; tc < 8; ++tc) {
                V16 Bh, Bl;
                load_frag(w2k + (size_t)((nbase >> 4) + tc) * TILE_DW, lane, Bh, Bl);
#pragma unroll
                for (int r = 0; r < 4; ++r) {
                    acc[r][tc] = wmma_bf16(Ah[r], Bh, acc[r][tc]);
                    acc[r][tc] = wmma_bf16(Ah[r], Bl, acc[r][tc]);
                    acc[r][tc] = wmma_bf16(Al[r], Bh, acc[r][tc]);
                }
            }
        }
        __syncthreads();
    }

    // ---- epilogue: + b2, store y ----
#pragma unroll
    for (int tc = 0; tc < 8; ++tc) {
        const int   col = nbase + tc * 16 + (lane & 15);
        const float bv  = b2e[col];
#pragma unroll
        for (int r = 0; r < 4; ++r) {
            const int mrow0 = rowbase + r * 16 + ((lane >> 4) << 3);
#pragma unroll
            for (int v = 0; v < 8; ++v)
                y[(size_t)(mrow0 + v) * D_ + col] = acc[r][tc][v] + bv;
        }
    }
}

// ================= fallback (no workspace): convert in-flight, TM=32 =================

__device__ __forceinline__ void load_a_f32(const float* __restrict__ src, int ld, int k0,
                                           int lane, V16& ah, V16& al) {
    const int m  = lane & 15;
    const int kh = (lane >> 4) << 3;
    const float* row = src + m * ld + k0;
#pragma unroll
    for (int j = 0; j < 8; ++j) {
        int koff = ((j & 4) << 2) + kh + ((j & 3) << 1);
        split2(row[koff], row[koff + 1], ah.u[j], al.u[j]);
    }
}

__device__ __forceinline__ void load_b_f32(const float* __restrict__ src, int ld, int k0, int n0,
                                           int lane, V16& bh, V16& bl) {
    const int n  = lane & 15;
    const int kh = (lane >> 4) << 3;
    const float* col = src + (size_t)(k0 + kh) * ld + n0 + n;
#pragma unroll
    for (int j = 0; j < 8; ++j) {
        int kj = ((j & 4) << 2) + ((j & 3) << 1);
        split2(col[(size_t)kj * ld], col[(size_t)(kj + 1) * ld], bh.u[j], bl.u[j]);
    }
}

__global__ __launch_bounds__(256, 1)
void moe_ffn_fused_fallback(const float* __restrict__ x,
                            const float* __restrict__ w1,
                            const float* __restrict__ b1,
                            const float* __restrict__ w2,
                            const float* __restrict__ b2,
                            float* __restrict__ y)
{
    __shared__ uint32_t lds_h[TMF * FK];

    const int wg      = blockIdx.x;
    const int e       = wg >> 6;
    const int rt      = wg & 63;
    const int rowbase = e * C_ + rt * TMF;

    const int tid  = threadIdx.x;
    const int lane = tid & 31;
    const int w    = tid >> 5;

    const float* xe  = x  + (size_t)rowbase * D_;
    const float* w1e = w1 + (size_t)e * D_ * F_;
    const float* b1e = b1 + (size_t)e * F_;
    const float* w2e = w2 + (size_t)e * F_ * D_;
    const float* b2e = b2 + (size_t)e * D_;

    const int hr = w >> 2;
    const int hc = w & 3;
    const int nbase = w * 128;

    v8f acc[2][8];
#pragma unroll
    for (int r = 0; r < 2; ++r)
#pragma unroll
        for (int t = 0; t < 8; ++t)
            acc[r][t] = (v8f){0.f, 0.f, 0.f, 0.f, 0.f, 0.f, 0.f, 0.f};

#pragma unroll 1
    for (int fc = 0; fc < F_ / FK; ++fc) {
        const int f0 = fc * FK;

        v8f h0 = (v8f){0.f, 0.f, 0.f, 0.f, 0.f, 0.f, 0.f, 0.f};
        v8f h1 = h0, h2 = h0;
#pragma unroll 2
        for (int ks = 0; ks < D_ / 32; ++ks) {
            V16 ah, al, bh, bl;
            load_a_f32(xe + (size_t)hr * 16 * D_, D_, ks * 32, lane, ah, al);
            load_b_f32(w1e, F_, ks * 32, f0 + hc * 16, lane, bh, bl);
            h0 = wmma_bf16(ah, bh, h0);
            h1 = wmma_bf16(ah, bl, h1);
            h2 = wmma_bf16(al, bh, h2);
        }
        v8f hacc = h0 + h1 + h2;
        {
            const float bv    = b1e[f0 + hc * 16 + (lane & 15)];
            const int   mrow0 = hr * 16 + ((lane >> 4) << 3);
            const int   col   = hc * 16 + (lane & 15);
#pragma unroll
            for (int v = 0; v < 8; ++v)
                lds_h[(mrow0 + v) * FK + col] = pack_split(gelu_tanh(hacc[v] + bv));
        }
        __syncthreads();

#pragma unroll
        for (int kk = 0; kk < 2; ++kk) {
            V16 Ah[2], Al[2];
#pragma unroll
            for (int r = 0; r < 2; ++r)
                load_a_lds(lds_h + r * 16 * FK, FK, kk * 32, lane, Ah[r], Al[r]);
#pragma unroll
            for (int tc = 0; tc < 8; ++tc) {
                V16 Bh, Bl;
                load_b_f32(w2e, D_, f0 + kk * 32, nbase + tc * 16, lane, Bh, Bl);
#pragma unroll
                for (int r = 0; r < 2; ++r) {
                    acc[r][tc] = wmma_bf16(Ah[r], Bh, acc[r][tc]);
                    acc[r][tc] = wmma_bf16(Ah[r], Bl, acc[r][tc]);
                    acc[r][tc] = wmma_bf16(Al[r], Bh, acc[r][tc]);
                }
            }
        }
        __syncthreads();
    }

#pragma unroll
    for (int tc = 0; tc < 8; ++tc) {
        const int   col = nbase + tc * 16 + (lane & 15);
        const float bv  = b2e[col];
#pragma unroll
        for (int r = 0; r < 2; ++r) {
            const int mrow0 = rowbase + r * 16 + ((lane >> 4) << 3);
#pragma unroll
            for (int v = 0; v < 8; ++v)
                y[(size_t)(mrow0 + v) * D_ + col] = acc[r][tc][v] + bv;
        }
    }
}

extern "C" void kernel_launch(void* const* d_in, const int* in_sizes, int n_in,
                              void* d_out, int out_size, void* d_ws, size_t ws_size,
                              hipStream_t stream) {
    (void)in_sizes; (void)n_in; (void)out_size;
    const float* x  = (const float*)d_in[0];
    const float* w1 = (const float*)d_in[1];
    const float* b1 = (const float*)d_in[2];
    const float* w2 = (const float*)d_in[3];
    const float* b2 = (const float*)d_in[4];
    float* y = (float*)d_out;

    dim3 block(256);

    if (ws_size >= WS_NEED_BYTES) {
        uint32_t* xs  = (uint32_t*)d_ws;
        uint32_t* w1s = xs + XS_DW;
        uint32_t* w2s = xs + XS_DW + W1S_DW;

        {   // swizzle x
            size_t n = (size_t)(T_ / 16) * (D_ / 32) * 256;
            hipLaunchKernelGGL(swizzle_x_kernel, dim3((n + 255) / 256), block, 0, stream, x, xs);
        }
        {   // swizzle w1 (K=D, N=F)
            size_t n = (size_t)E_ * (D_ / 32) * (F_ / 16) * 256;
            hipLaunchKernelGGL(swizzle_w_kernel, dim3((n + 255) / 256), block, 0, stream,
                               w1, w1s, D_, F_);
        }
        {   // swizzle w2 (K=F, N=D)
            size_t n = (size_t)E_ * (F_ / 32) * (D_ / 16) * 256;
            hipLaunchKernelGGL(swizzle_w_kernel, dim3((n + 255) / 256), block, 0, stream,
                               w2, w2s, F_, D_);
        }
        hipLaunchKernelGGL(moe_ffn_fused_sw, dim3(E_ * (C_ / TMS)), block, 0, stream,
                           xs, w1s, b1, w2s, b2, y);          // 256 workgroups
    } else {
        hipLaunchKernelGGL(moe_ffn_fused_fallback, dim3(E_ * (C_ / TMF)), block, 0, stream,
                           x, w1, b1, w2, b2, y);             // 512 workgroups
    }
}
